// Contrastive_18159121727571
// MI455X (gfx1250) — matrix operand order repaired and checked by reading, not compiled, not run on hardware
//
#include <hip/hip_runtime.h>

typedef __attribute__((ext_vector_type(16))) _Float16 v16h;
typedef __attribute__((ext_vector_type(8)))  float    v8f;

#define B_SEQ 24
#define L_SEQ 512
#define D_IN  64
#define H_DIM 64
#define HJ_STRIDE 72   // halves; 144B pitch = 9*16B -> 16B-aligned & LDS-bank-conflict-free

union FragU { uint4 u4[2]; v16h h; };

__device__ __forceinline__ float gelu_tanh(float v) {
    float u = 0.7978845608028654f * (v + 0.044715f * v * v * v);
    return 0.5f * v * (1.0f + tanhf(u));
}

// ---------------- Kernel 1: h = gelu(x @ W + b) -> f16, row-major [B*L, H] -------------
__global__ void encode_gelu_f16(const float* __restrict__ x,
                                const float* __restrict__ W,
                                const float* __restrict__ bias,
                                _Float16* __restrict__ h) {
    __shared__ float Wl[D_IN * H_DIM];   // 16 KB
    __shared__ float xl[4 * D_IN];       // 4 rows per block
    const int tid = threadIdx.x;
    for (int k = tid; k < D_IN * H_DIM; k += 256) Wl[k] = W[k];
    xl[tid] = x[(size_t)blockIdx.x * 256 + tid];
    __syncthreads();
    const int r   = tid >> 6;   // row within block (0..3)
    const int col = tid & 63;
    const float* xr = xl + r * D_IN;
    float acc = bias[col];
    #pragma unroll
    for (int d = 0; d < D_IN; ++d) acc = fmaf(xr[d], Wl[d * H_DIM + col], acc);
    const size_t row = (size_t)blockIdx.x * 4 + r;
    h[row * H_DIM + col] = (_Float16)gelu_tanh(acc);
}

// ------- Kernel 2: per pair (i<=j): logsumexp over h_i @ h_j^T via WMMA, fused --------
__global__ void pair_lse_wmma(const _Float16* __restrict__ h,
                              float* __restrict__ out) {
    // decode flat upper-triangular pair index -> (i, j), i <= j
    int t = blockIdx.x, i = 0;
    while (t >= (B_SEQ - i)) { t -= (B_SEQ - i); ++i; }
    const int j = i + t;

    extern __shared__ char smem_raw[];
    _Float16* hj = (_Float16*)smem_raw;                                  // 512 x 72 halves
    float* red   = (float*)(smem_raw + L_SEQ * HJ_STRIDE * sizeof(_Float16));

    const _Float16* hig = h + (size_t)i * L_SEQ * H_DIM;
    const _Float16* hjg = h + (size_t)j * L_SEQ * H_DIM;

    const int tid = threadIdx.x;
    // stage h_j into padded LDS, 16B chunks (row pitch 144B is 16B-aligned)
    for (int c = tid; c < L_SEQ * (H_DIM / 8); c += 256) {
        const int row = c >> 3, cc = c & 7;
        *((uint4*)(hj + (size_t)row * HJ_STRIDE) + cc) = ((const uint4*)hjg)[c];
    }
    __syncthreads();

    const int wave = tid >> 5, lane = tid & 31;
    const int mrow = lane & 15;                 // M (A) / N (B) index for this lane
    const int kbA  = (lane < 16) ? 0 : 16;      // contiguous A-frag half-wave K base (halves)
    const int kbB  = (lane < 16) ? 0 : 16;      // contiguous B-frag half-wave K base (halves)

    float m = -3.402823466e38f;   // running max
    float s = 0.0f;               // running sum of exp(x - m)

    // 16x16 tiles of the 512x512 score matrix; each wave owns 4 tile-rows.
    for (int tm = wave; tm < L_SEQ / 16; tm += 8) {
        const _Float16* ap = hig + (size_t)(tm * 16 + mrow) * H_DIM + kbA;
        FragU a0, a1;                              // A fragments: K 0..31 and 32..63
        a0.u4[0] = *(const uint4*)(ap);            // K kb..kb+7
        a0.u4[1] = *(const uint4*)(ap + 8);        // K kb+8..kb+15
        a1.u4[0] = *(const uint4*)(ap + 32);
        a1.u4[1] = *(const uint4*)(ap + 40);

        for (int tn = 0; tn < L_SEQ / 16; ++tn) {
            const _Float16* bp = hj + (size_t)(tn * 16 + mrow) * HJ_STRIDE + kbB;
            FragU b0, b1;                          // B fragments: contiguous K halves
            b0.u4[0] = *(const uint4*)(bp);
            b0.u4[1] = *(const uint4*)(bp + 8);
            b1.u4[0] = *(const uint4*)(bp + 32);
            b1.u4[1] = *(const uint4*)(bp + 40);

            v8f c = {};
            c = __builtin_amdgcn_wmma_f32_16x16x32_f16(false, a0.h, false, b0.h,
                                                       (short)0, c, false, false);
            c = __builtin_amdgcn_wmma_f32_16x16x32_f16(false, a1.h, false, b1.h,
                                                       (short)0, c, false, false);

            // online logsumexp over the 8 accumulator elements of this lane
            float t0 = fmaxf(fmaxf(c[0], c[1]), fmaxf(c[2], c[3]));
            float t1 = fmaxf(fmaxf(c[4], c[5]), fmaxf(c[6], c[7]));
            float nm = fmaxf(m, fmaxf(t0, t1));
            s *= __expf(m - nm);
            s += __expf(c[0] - nm) + __expf(c[1] - nm) + __expf(c[2] - nm) + __expf(c[3] - nm)
               + __expf(c[4] - nm) + __expf(c[5] - nm) + __expf(c[6] - nm) + __expf(c[7] - nm);
            m = nm;
        }
    }

    // reduce (m, s) across the 32 lanes of the wave
    for (int off = 16; off > 0; off >>= 1) {
        float om = __shfl_xor(m, off, 32);
        float os = __shfl_xor(s, off, 32);
        float nm = fmaxf(m, om);
        s = s * __expf(m - nm) + os * __expf(om - nm);
        m = nm;
    }
    if (lane == 0) { red[wave * 2] = m; red[wave * 2 + 1] = s; }
    __syncthreads();

    if (tid == 0) {
        float M = red[0], S = red[1];
        for (int w = 1; w < 8; ++w) {
            float om = red[w * 2], os = red[w * 2 + 1];
            float nm = fmaxf(M, om);
            S = S * __expf(M - nm) + os * __expf(om - nm);
            M = nm;
        }
        const float r = M + logf(S);
        out[i * B_SEQ + j] = r;
        out[j * B_SEQ + i] = r;   // scores are symmetric in (i, j)
    }
}

extern "C" void kernel_launch(void* const* d_in, const int* in_sizes, int n_in,
                              void* d_out, int out_size, void* d_ws, size_t ws_size,
                              hipStream_t stream) {
    const float* x = (const float*)d_in[0];   // [24, 512, 64]
    const float* W = (const float*)d_in[1];   // [64, 64]
    const float* b = (const float*)d_in[2];   // [64]
    float* out = (float*)d_out;               // [24, 24]
    _Float16* hbuf = (_Float16*)d_ws;         // [24*512, 64] f16 (1.5 MB)

    encode_gelu_f16<<<(B_SEQ * L_SEQ) / 4, 256, 0, stream>>>(x, W, b, hbuf);

    const size_t shmem = L_SEQ * HJ_STRIDE * sizeof(_Float16) + 16 * sizeof(float);
    pair_lse_wmma<<<(B_SEQ * (B_SEQ + 1)) / 2, 256, shmem, stream>>>(hbuf, out);
}
